// GATv2MolecularPredictor_49546742726913
// MI455X (gfx1250) — compile-verified, matrix operand order checked by reading, not run on hardware
//
#include <hip/hip_runtime.h>
#include <hip/hip_bf16.h>

// GATv2 molecular predictor for MI455X (gfx1250, wave32, WMMA).
// Dense layers: f16-in / f32-acc via v_wmma_f32_16x16x32_f16. All GEMMs here
// are HBM-bandwidth-bound (23.3 TB/s), so activations stream as b128 loads and
// weights are pre-packed into WMMA fragment order (contiguous 32B per lane per
// k-tile) so the B path is pure global_load_b128. K is a template parameter so
// the k-loop fully unrolls (static v_wmma chain, software-pipelined loads).
// Scatter phases (attention softmax, segment means): wave32 shfl reductions +
// float atomics, ordered-uint atomicMax for segment max.

typedef _Float16 half_t;
typedef __attribute__((ext_vector_type(16))) _Float16 v16h;
typedef __attribute__((ext_vector_type(8)))  float    v8f;
typedef __attribute__((ext_vector_type(4)))  float    v4f;

#define HIDF 128
#define HEADS 4
#define DJK 512

// ---------------- weight pre-pack: f32 [K x Nc] -> f16 WMMA fragments -------
// Packed layout: tile t = nt*(Kpad/32)+kt holds a 32x16 B fragment as
// P[t*512 + lane*16 + elem]; elem e=2v+p maps to W[kt*32 + kfrag(v,khalf)+p][nt*16+(lane&15)]
// with kfrag = ((v&3)<<1)+(khalf<<3)+((v>>2)<<4)  (ISA 7.12.2 16-bit 32x16 layout).
// Rows kk >= Krows pack as zero (K padding for Ka < 32 cases).
__global__ void k_pack_w(const float* __restrict__ W, half_t* __restrict__ P,
                         int Krows, int Kpad, int Nc) {
  int i = blockIdx.x * blockDim.x + threadIdx.x;
  if (i >= Kpad * Nc) return;
  int elem = i & 15;
  int lane = (i >> 4) & 31;
  int tile = i >> 9;
  int KT = Kpad >> 5;
  int nt = tile / KT;
  int kt = tile - nt * KT;
  int v = elem >> 1, p = elem & 1;
  int khalf = lane >> 4;
  int n = nt * 16 + (lane & 15);
  int kk = kt * 32 + ((v & 3) << 1) + (khalf << 3) + ((v >> 2) << 4) + p;
  P[i] = (kk < Krows) ? (half_t)W[(size_t)kk * Nc + n] : (half_t)0.f;
}

__global__ void k_fill_u32(unsigned* __restrict__ p, unsigned v, int n) {
  int i = blockIdx.x * blockDim.x + threadIdx.x;
  if (i < n) p[i] = v;
}

// ---------------- WMMA GEMM: C = act(bn(A@W + bias)) ------------------------
// A: [M x >=Ka] f32, leading dim lda. Out-of-range K columns are only address-
// clamped (values harmless: packed B rows there are zero). Invalid M rows are
// clamped to row 0 (their C rows are never stored). Optional fused row-add:
// A_eff[m,k] = A[m,k] + Aadd[arow[m]*K + k] (only used with Ka == K).
template <int K>
__global__ void k_gemm(const float* __restrict__ A, int lda, int Ka,
                       const half_t* __restrict__ Wp,
                       const float* __restrict__ bias,
                       const float* __restrict__ bng, const float* __restrict__ bnb,
                       const float* __restrict__ Aadd, const int* __restrict__ arow,
                       float* __restrict__ C, int ldc, int M, int Nc, int act) {
  constexpr int KT = K / 32;
  const int lane  = threadIdx.x;                 // 0..31
  const int m0    = (blockIdx.x * blockDim.y + threadIdx.y) * 16;
  const int khalf = lane >> 4;
  const int mr    = m0 + (lane & 15);
  const bool mvalid = (mr < M);
  const float*  Ap = A + (size_t)(mvalid ? mr : 0) * lda;
  const float*  Rp = Aadd ? (Aadd + (size_t)(mvalid ? arow[mr] : 0) * K) : nullptr;
  const half_t* Bp = Wp + (size_t)blockIdx.y * KT * 512 + lane * 16;

  v8f acc = {};
#pragma unroll
  for (int kt = 0; kt < KT; ++kt) {
    const int b1 = kt * 32 + (khalf << 3);       // af[0..7]  = A[b1 .. b1+7]
    const int q0 = b1, q1 = b1 + 4, q2 = b1 + 16, q3 = b1 + 20;
    v4f a0 = *(const v4f*)(Ap + (q0 < Ka ? q0 : 0));
    v4f a1 = *(const v4f*)(Ap + (q1 < Ka ? q1 : 0));
    v4f a2 = *(const v4f*)(Ap + (q2 < Ka ? q2 : 0));  // af[8..15] = A[b1+16 .. b1+23]
    v4f a3 = *(const v4f*)(Ap + (q3 < Ka ? q3 : 0));
    if (Rp) {
      a0 += *(const v4f*)(Rp + q0);  a1 += *(const v4f*)(Rp + q1);
      a2 += *(const v4f*)(Rp + q2);  a3 += *(const v4f*)(Rp + q3);
    }
    v16h af;
#pragma unroll
    for (int j = 0; j < 4; ++j) {
      af[j]      = (half_t)a0[j];
      af[4 + j]  = (half_t)a1[j];
      af[8 + j]  = (half_t)a2[j];
      af[12 + j] = (half_t)a3[j];
    }
    v16h bf = *(const v16h*)(Bp + (size_t)kt * 512);   // 32B: 2x global_load_b128
    acc = __builtin_amdgcn_wmma_f32_16x16x32_f16(false, af, false, bf,
                                                 (short)0, acc, false, false);
  }

  const int nc = blockIdx.y * 16 + (lane & 15);
  float bv = bias ? bias[nc] : 0.f;
  float g = 1.f, bb = 0.f;
  if (bng) { g = bng[nc] * 0.99999500003749971f; bb = bnb[nc]; }  // 1/sqrt(1+1e-5)
#pragma unroll
  for (int r = 0; r < 8; ++r) {
    int row = m0 + r + (khalf << 3);    // C layout: VGPR r -> M=r (+8 hi lanes)
    if (row < M) {
      float t = acc[r] + bv;
      if (bng) t = t * g + bb;
      if (act == 1) t = fmaxf(t, 0.f);
      C[(size_t)row * ldc + nc] = t;
    }
  }
}

// ---------------- segment ops ----------------
__global__ void k_seg_count(const int* __restrict__ seg, float* __restrict__ cnt, int rows) {
  int i = blockIdx.x * blockDim.x + threadIdx.x;
  if (i < rows) atomicAdd(&cnt[seg[i]], 1.f);
}
__global__ void k_seg_accum(const float* __restrict__ src, const int* __restrict__ seg,
                            const float* __restrict__ gate, float* __restrict__ sum,
                            int rows, int F) {
  int r = blockIdx.x;
  if (r >= rows) return;
  int g = seg[r];
  float gv = gate ? gate[r] : 1.f;
  for (int j = threadIdx.x; j < F; j += blockDim.x)
    atomicAdd(&sum[(size_t)g * F + j], src[(size_t)r * F + j] * gv);
}
__global__ void k_seg_div(float* __restrict__ sum, const float* __restrict__ cnt,
                          int nseg, int logF) {
  int i = blockIdx.x * blockDim.x + threadIdx.x;
  if (i < (nseg << logF)) sum[i] /= fmaxf(cnt[i >> logF], 1.f);
}

// ---------------- GATv2 attention ----------------
__device__ __forceinline__ unsigned ford(float f) {
  unsigned u = __float_as_uint(f);
  return (u & 0x80000000u) ? ~u : (u | 0x80000000u);
}
__device__ __forceinline__ float fdec(unsigned u) {
  return __uint_as_float((u & 0x80000000u) ? (u ^ 0x80000000u) : ~u);
}

__global__ void k_attn_score(const float* __restrict__ xl, const float* __restrict__ xr,
                             const float* __restrict__ ee, const int* __restrict__ ei,
                             const float* __restrict__ att, float* __restrict__ s,
                             int E, int Etot) {
  int e = blockIdx.x;
  if (e >= Etot) return;
  int sN, dN;
  if (e < E) { sN = ei[e]; dN = ei[E + e]; } else { sN = dN = e - E; }
  int c = threadIdx.x;                       // 128 threads; one wave == one head
  float m = xl[(size_t)sN * HIDF + c] + xr[(size_t)dN * HIDF + c] + ee[(size_t)e * HIDF + c];
  m = (m > 0.f) ? m : 0.2f * m;              // leaky_relu 0.2
  float p = m * att[c];
#pragma unroll
  for (int o = 16; o; o >>= 1) p += __shfl_xor(p, o, 32);
  if ((c & 31) == 0) s[(size_t)e * HEADS + (c >> 5)] = p;
}
__global__ void k_attn_max(const float* __restrict__ s, const int* __restrict__ ei,
                           unsigned* __restrict__ smax, int E, int Etot) {
  int i = blockIdx.x * blockDim.x + threadIdx.x;
  if (i >= Etot * HEADS) return;
  int e = i >> 2, h = i & 3;
  int dN = (e < E) ? ei[E + e] : (e - E);
  atomicMax(&smax[(size_t)dN * HEADS + h], ford(s[i]));
}
__global__ void k_attn_expden(float* __restrict__ s, const int* __restrict__ ei,
                              const unsigned* __restrict__ smax, float* __restrict__ den,
                              int E, int Etot) {
  int i = blockIdx.x * blockDim.x + threadIdx.x;
  if (i >= Etot * HEADS) return;
  int e = i >> 2, h = i & 3;
  int dN = (e < E) ? ei[E + e] : (e - E);
  float ex = expf(s[i] - fdec(smax[(size_t)dN * HEADS + h]));
  s[i] = ex;
  atomicAdd(&den[(size_t)dN * HEADS + h], ex);
}
__global__ void k_attn_accum(const float* __restrict__ s, const float* __restrict__ den,
                             const float* __restrict__ xl, const int* __restrict__ ei,
                             float* __restrict__ tacc, int E, int Etot) {
  int e = blockIdx.x;
  if (e >= Etot) return;
  int sN, dN;
  if (e < E) { sN = ei[e]; dN = ei[E + e]; } else { sN = dN = e - E; }
  int c = threadIdx.x, h = c >> 5;
  float alpha = s[(size_t)e * HEADS + h] / (den[(size_t)dN * HEADS + h] + 1e-16f);
  atomicAdd(&tacc[(size_t)dN * HIDF + c], xl[(size_t)sN * HIDF + c] * alpha);
}

// ---------------- LayerNorm + ReLU + residual -> xj slice ----------------
__global__ void k_ln_res(const float* __restrict__ tacc, const float* __restrict__ cb,
                         const float* __restrict__ g, const float* __restrict__ b,
                         const float* __restrict__ res, int resld,
                         float* __restrict__ out, int outld, int N) {
  int n = blockIdx.x;
  if (n >= N) return;
  int c = threadIdx.x;
  float v = tacc[(size_t)n * HIDF + c] + cb[c];
  float s1 = v, s2 = v * v;
#pragma unroll
  for (int o = 16; o; o >>= 1) { s1 += __shfl_xor(s1, o, 32); s2 += __shfl_xor(s2, o, 32); }
  __shared__ float red[8];
  int w = c >> 5;
  if ((c & 31) == 0) { red[w] = s1; red[4 + w] = s2; }
  __syncthreads();
  float mean = (red[0] + red[1] + red[2] + red[3]) * (1.f / HIDF);
  float e2   = (red[4] + red[5] + red[6] + red[7]) * (1.f / HIDF);
  float var  = e2 - mean * mean;
  float y = (v - mean) * rsqrtf(var + 1e-5f) * g[c] + b[c];
  y = fmaxf(y, 0.f) + res[(size_t)n * resld + c];
  out[(size_t)n * outld + c] = y;
}

// ---------------- row dot (M x K @ K x 1), optional sigmoid ----------------
__global__ void k_rowdot(const float* __restrict__ A, const float* __restrict__ w,
                         const float* __restrict__ b, float* __restrict__ out,
                         int M, int K, int act) {
  int lane = threadIdx.x & 31;
  int row = blockIdx.x * (blockDim.x >> 5) + (threadIdx.x >> 5);
  if (row >= M) return;
  const float* a = A + (size_t)row * K;
  float p = 0.f;
  for (int k = lane; k < K; k += 32) p += a[k] * w[k];
#pragma unroll
  for (int o = 16; o; o >>= 1) p += __shfl_xor(p, o, 32);
  if (lane == 0) {
    float t = p + b[0];
    if (act == 1) t = 1.f / (1.f + expf(-t));
    out[row] = t;
  }
}

// ---------------- GRU elementwise update (in-place gr) ----------------
__global__ void k_gru(const float* __restrict__ ggi, const float* __restrict__ ggh,
                      float* __restrict__ gr, int G) {
  int i = blockIdx.x * blockDim.x + threadIdx.x;
  if (i >= G * DJK) return;
  int g = i / DJK, j = i % DJK;
  size_t b3 = (size_t)g * 3 * DJK;
  float ir = ggi[b3 + j], iz = ggi[b3 + DJK + j], inn = ggi[b3 + 2 * DJK + j];
  float hr = ggh[b3 + j], hz = ggh[b3 + DJK + j], hnn = ggh[b3 + 2 * DJK + j];
  float r = 1.f / (1.f + expf(-(ir + hr)));
  float z = 1.f / (1.f + expf(-(iz + hz)));
  float cand = tanhf(inn + r * hnn);
  gr[i] = fmaxf((1.f - z) * cand + z * gr[i], 0.f);
}

// ---------------- host orchestration ----------------
extern "C" void kernel_launch(void* const* d_in, const int* in_sizes, int n_in,
                              void* d_out, int out_size, void* d_ws, size_t ws_size,
                              hipStream_t stream) {
  (void)n_in; (void)ws_size;
  const int N = in_sizes[0] / 64;     // 100000
  const int E = in_sizes[1] / 16;     // 200000
  const int G = out_size;             // 4096
  const int Etot = E + N;

  const float* x         = (const float*)d_in[0];
  const float* edge_attr = (const float*)d_in[1];
  const int*   ei        = (const int*)  d_in[2];
  const int*   batch     = (const int*)  d_in[3];
  const float* node_W = (const float*)d_in[4];  const float* node_b = (const float*)d_in[5];
  const float* edge_W = (const float*)d_in[6];  const float* edge_b = (const float*)d_in[7];
  const float* conv_Wl = (const float*)d_in[8];  const float* conv_bl = (const float*)d_in[9];
  const float* conv_Wr = (const float*)d_in[10]; const float* conv_br = (const float*)d_in[11];
  const float* conv_We = (const float*)d_in[12]; const float* conv_att = (const float*)d_in[13];
  const float* conv_bias = (const float*)d_in[14];
  const float* ln_g = (const float*)d_in[15]; const float* ln_b = (const float*)d_in[16];
  const float* gate_W1 = (const float*)d_in[17]; const float* gate_b1 = (const float*)d_in[18];
  const float* gate_W2 = (const float*)d_in[19]; const float* gate_b2 = (const float*)d_in[20];
  const float* gru_Wih = (const float*)d_in[21]; const float* gru_bih = (const float*)d_in[22];
  const float* gru_Whh = (const float*)d_in[23]; const float* gru_bhh = (const float*)d_in[24];
  const float* p1_W = (const float*)d_in[25]; const float* p1_b = (const float*)d_in[26];
  const float* bn1_g = (const float*)d_in[27]; const float* bn1_b = (const float*)d_in[28];
  const float* p2_W = (const float*)d_in[29]; const float* p2_b = (const float*)d_in[30];
  const float* bn2_g = (const float*)d_in[31]; const float* bn2_b = (const float*)d_in[32];
  const float* p3_W = (const float*)d_in[33]; const float* p3_b = (const float*)d_in[34];
  const float* bn3_g = (const float*)d_in[35]; const float* bn3_b = (const float*)d_in[36];
  const float* p4_W = (const float*)d_in[37]; const float* p4_b = (const float*)d_in[38];

  char* ws = (char*)d_ws;
  size_t off = 0;
  auto alloc = [&](size_t bytes) -> char* {
    size_t o = (off + 255) & ~(size_t)255; off = o + bytes; return ws + o;
  };

  // --- persistent allocations ---
  half_t* w_node = (half_t*)alloc((size_t)64 * 128 * 2);
  half_t* w_edge = (half_t*)alloc((size_t)32 * 128 * 2);   // K padded 16->32
  half_t *w_l[4], *w_r[4], *w_e[4];
  for (int i = 0; i < 4; i++) w_l[i] = (half_t*)alloc((size_t)128 * 128 * 2);
  for (int i = 0; i < 4; i++) w_r[i] = (half_t*)alloc((size_t)128 * 128 * 2);
  for (int i = 0; i < 4; i++) w_e[i] = (half_t*)alloc((size_t)128 * 128 * 2);
  half_t* w_g1 = (half_t*)alloc((size_t)512 * 512 * 2);
  half_t* w_ih = (half_t*)alloc((size_t)512 * 1536 * 2);
  half_t* w_hh = (half_t*)alloc((size_t)512 * 1536 * 2);
  half_t* w_p1 = (half_t*)alloc((size_t)512 * 256 * 2);
  half_t* w_p2 = (half_t*)alloc((size_t)256 * 128 * 2);
  half_t* w_p3 = (half_t*)alloc((size_t)128 * 64 * 2);

  float*    xj   = (float*)alloc((size_t)N * DJK * 4);
  float*    gr   = (float*)alloc((size_t)G * DJK * 4);
  float*    newr = (float*)alloc((size_t)G * DJK * 4);
  float*    ggi  = (float*)alloc((size_t)G * 3 * DJK * 4);
  float*    ggh  = (float*)alloc((size_t)G * 3 * DJK * 4);
  float*    gate = (float*)alloc((size_t)N * 4);
  float*    gcnt = (float*)alloc((size_t)G * 4);
  float*    ncnt = (float*)alloc((size_t)N * 4);
  float*    sbuf = (float*)alloc((size_t)Etot * HEADS * 4);
  unsigned* smax = (unsigned*)alloc((size_t)N * HEADS * 4);
  float*    den  = (float*)alloc((size_t)N * HEADS * 4);
  float*    o1   = (float*)alloc((size_t)G * 256 * 4);
  float*    o2   = (float*)alloc((size_t)G * 128 * 4);
  float*    o3   = (float*)alloc((size_t)G * 64 * 4);

  // --- conv-phase region (reused as t1 during pooling) ---
  float* h0      = (float*)alloc((size_t)N * HIDF * 4);
  float* ea_full = (float*)alloc((size_t)Etot * HIDF * 4);
  float* ee      = (float*)alloc((size_t)Etot * HIDF * 4);
  float* xl      = (float*)alloc((size_t)N * HIDF * 4);
  float* xr      = (float*)alloc((size_t)N * HIDF * 4);
  float* tacc    = (float*)alloc((size_t)N * HIDF * 4);
  float* t1      = h0;   // overlay: N*512*4 (205MB) fits in conv region (512MB)

  auto pack = [&](const float* W, half_t* P, int Krows, int Kpad, int Nc) {
    int tot = Kpad * Nc;
    k_pack_w<<<dim3((tot + 255) / 256), dim3(256), 0, stream>>>(W, P, Krows, Kpad, Nc);
  };
  auto gemm = [&](const float* A, int lda, int Ka, const half_t* W, const float* bias,
                  const float* bng, const float* bnb, const float* Aadd, const int* arow,
                  float* C, int ldc, int M, int Nc, int K, int act) {
    dim3 blk(32, 4), grd((M + 63) / 64, Nc / 16);
    switch (K) {
      case 32:  k_gemm<32><<<grd, blk, 0, stream>>>(A, lda, Ka, W, bias, bng, bnb, Aadd, arow, C, ldc, M, Nc, act); break;
      case 64:  k_gemm<64><<<grd, blk, 0, stream>>>(A, lda, Ka, W, bias, bng, bnb, Aadd, arow, C, ldc, M, Nc, act); break;
      case 128: k_gemm<128><<<grd, blk, 0, stream>>>(A, lda, Ka, W, bias, bng, bnb, Aadd, arow, C, ldc, M, Nc, act); break;
      case 256: k_gemm<256><<<grd, blk, 0, stream>>>(A, lda, Ka, W, bias, bng, bnb, Aadd, arow, C, ldc, M, Nc, act); break;
      default:  k_gemm<512><<<grd, blk, 0, stream>>>(A, lda, Ka, W, bias, bng, bnb, Aadd, arow, C, ldc, M, Nc, act); break;
    }
  };

  // ---- weight pre-pack into WMMA fragment order (once per call) ----
  pack(node_W, w_node, 64, 64, 128);
  pack(edge_W, w_edge, 16, 32, 128);     // zero-padded K rows 16..31
  for (int i = 0; i < 4; i++) {
    pack(conv_Wl + (size_t)i * 16384, w_l[i], 128, 128, 128);
    pack(conv_Wr + (size_t)i * 16384, w_r[i], 128, 128, 128);
    pack(conv_We + (size_t)i * 16384, w_e[i], 128, 128, 128);
  }
  pack(gate_W1, w_g1, 512, 512, 512);
  pack(gru_Wih, w_ih, 512, 512, 1536);
  pack(gru_Whh, w_hh, 512, 512, 1536);
  pack(p1_W, w_p1, 512, 512, 256);
  pack(p2_W, w_p2, 256, 256, 128);
  pack(p3_W, w_p3, 128, 128, 64);

  // ---- encoders ----
  gemm(x, 64, 64, w_node, node_b, 0, 0, 0, 0, h0, 128, N, 128, 64, 1);
  gemm(edge_attr, 16, 16, w_edge, edge_b, 0, 0, 0, 0, ea_full, 128, E, 128, 32, 1);

  // self-loop edge features = seg_mean(ea, dst0) -> ea_full[E:]
  hipMemsetAsync(ea_full + (size_t)E * HIDF, 0, (size_t)N * HIDF * 4, stream);
  hipMemsetAsync(ncnt, 0, (size_t)N * 4, stream);
  k_seg_count<<<dim3((E + 255) / 256), dim3(256), 0, stream>>>(ei + E, ncnt, E);
  k_seg_accum<<<dim3(E), dim3(128), 0, stream>>>(ea_full, ei + E, nullptr,
                                                 ea_full + (size_t)E * HIDF, E, HIDF);
  k_seg_div<<<dim3((N * HIDF + 255) / 256), dim3(256), 0, stream>>>(
      ea_full + (size_t)E * HIDF, ncnt, N, 7);

  // ---- 4 GATv2 layers ----
  for (int i = 0; i < 4; i++) {
    const float* hin = (i == 0) ? h0 : (xj + (size_t)(i - 1) * HIDF);
    int hld = (i == 0) ? HIDF : DJK;
    gemm(hin, hld, 128, w_l[i], conv_bl + i * 128, 0, 0, 0, 0, xl, 128, N, 128, 128, 0);
    gemm(hin, hld, 128, w_r[i], conv_br + i * 128, 0, 0, 0, 0, xr, 128, N, 128, 128, 0);
    gemm(ea_full, 128, 128, w_e[i], nullptr, 0, 0, 0, 0, ee, 128, Etot, 128, 128, 0);

    k_fill_u32<<<dim3((N * HEADS + 255) / 256), dim3(256), 0, stream>>>(
        smax, 0x007FFFFFu /* ford(-inf) */, N * HEADS);
    hipMemsetAsync(den, 0, (size_t)N * HEADS * 4, stream);

    k_attn_score<<<dim3(Etot), dim3(128), 0, stream>>>(xl, xr, ee, ei,
                                                       conv_att + i * 128, sbuf, E, Etot);
    k_attn_max<<<dim3((Etot * HEADS + 255) / 256), dim3(256), 0, stream>>>(sbuf, ei, smax, E, Etot);
    k_attn_expden<<<dim3((Etot * HEADS + 255) / 256), dim3(256), 0, stream>>>(sbuf, ei, smax, den, E, Etot);

    hipMemsetAsync(tacc, 0, (size_t)N * HIDF * 4, stream);
    k_attn_accum<<<dim3(Etot), dim3(128), 0, stream>>>(sbuf, den, xl, ei, tacc, E, Etot);

    k_ln_res<<<dim3(N), dim3(128), 0, stream>>>(tacc, conv_bias + i * 128,
                                                ln_g + i * 128, ln_b + i * 128,
                                                hin, hld, xj + (size_t)i * HIDF, DJK, N);
  }

  // ---- attentive pooling with GRU ----
  hipMemsetAsync(gr, 0, (size_t)G * DJK * 4, stream);
  hipMemsetAsync(gcnt, 0, (size_t)G * 4, stream);
  k_seg_count<<<dim3((N + 255) / 256), dim3(256), 0, stream>>>(batch, gcnt, N);
  k_seg_accum<<<dim3(N), dim3(256), 0, stream>>>(xj, batch, nullptr, gr, N, DJK);
  k_seg_div<<<dim3((G * DJK + 255) / 256), dim3(256), 0, stream>>>(gr, gcnt, G, 9);

  for (int it = 0; it < 2; ++it) {
    // t1 = relu((xj + gr[batch]) @ gate_W1 + b1)
    gemm(xj, DJK, DJK, w_g1, gate_b1, 0, 0, gr, batch, t1, DJK, N, DJK, DJK, 1);
    // gate = sigmoid(t1 @ gate_W2 + b2)
    k_rowdot<<<dim3((N + 7) / 8), dim3(256), 0, stream>>>(t1, gate_W2, gate_b2, gate, N, DJK, 1);
    // newr = seg_mean(xj * gate)
    hipMemsetAsync(newr, 0, (size_t)G * DJK * 4, stream);
    k_seg_accum<<<dim3(N), dim3(256), 0, stream>>>(xj, batch, gate, newr, N, DJK);
    k_seg_div<<<dim3((G * DJK + 255) / 256), dim3(256), 0, stream>>>(newr, gcnt, G, 9);
    // GRU gates
    gemm(newr, DJK, DJK, w_ih, gru_bih, 0, 0, 0, 0, ggi, 1536, G, 1536, DJK, 0);
    gemm(gr,   DJK, DJK, w_hh, gru_bhh, 0, 0, 0, 0, ggh, 1536, G, 1536, DJK, 0);
    k_gru<<<dim3((G * DJK + 255) / 256), dim3(256), 0, stream>>>(ggi, ggh, gr, G);
  }

  // ---- predictor MLP (eval-mode BN fused into GEMM epilogue) ----
  gemm(gr, DJK, DJK, w_p1, p1_b, bn1_g, bn1_b, 0, 0, o1, 256, G, 256, 512, 1);
  gemm(o1, 256, 256, w_p2, p2_b, bn2_g, bn2_b, 0, 0, o2, 128, G, 128, 256, 1);
  gemm(o2, 128, 128, w_p3, p3_b, bn3_g, bn3_b, 0, 0, o3, 64, G, 64, 128, 1);
  k_rowdot<<<dim3((G + 7) / 8), dim3(256), 0, stream>>>(o3, p4_W, p4_b, (float*)d_out, G, 64, 0);
}
